// SSFN_14096082665906
// MI455X (gfx1250) — matrix operand order repaired and checked by reading, not compiled, not run on hardware
//
#include <hip/hip_runtime.h>
#include <math.h>

typedef _Float16 f16;
typedef __attribute__((ext_vector_type(16))) _Float16 v16h;
typedef __attribute__((ext_vector_type(8)))  float    v8f;

#define C_DIM 128
#define M_DIM 512
#define IMG   256
#define NHEAD 4
#define HD    32
#define NTOK  64

static __device__ __forceinline__ v8f wmma_f16(v16h a, v16h b, v8f c) {
  return __builtin_amdgcn_wmma_f32_16x16x32_f16(false, a, false, b, (short)0, c, false, false);
}

// A fragment: 16x32 f16 tile at (m0,k0) from row-major src with leading dim ld.
static __device__ __forceinline__ v16h load_a(const f16* src, int ld, int m0, int k0, int lane) {
  int m  = m0 + (lane & 15);
  int hs = (lane >> 4) * 16;
  v16h a;
#pragma unroll
  for (int e = 0; e < 16; ++e) {
    int k = hs + e;
    a[e] = src[m * ld + k0 + k];
  }
  return a;
}

// B fragment (direct): B[k][n] = src[(k0+k)*ld + n0+n]
static __device__ __forceinline__ v16h load_b(const f16* src, int ld, int k0, int n0, int lane) {
  int n  = n0 + (lane & 15);
  int kb = k0 + (lane >> 4) * 16;
  v16h b;
#pragma unroll
  for (int e = 0; e < 16; ++e) b[e] = src[(kb + e) * ld + n];
  return b;
}

// B fragment (transposed weight): B[k][n] = W[n0+n][k0+k]
static __device__ __forceinline__ v16h load_bT(const f16* w, int ld, int n0, int k0, int lane) {
  int n  = n0 + (lane & 15);
  int kb = k0 + (lane >> 4) * 16;
  v16h b;
#pragma unroll
  for (int e = 0; e < 16; ++e) b[e] = w[n * ld + kb + e];
  return b;
}

static __device__ __forceinline__ float gelu_f(float x) {
  return 0.5f * x * (1.0f + erff(x * 0.70710678118654752f));
}
static __device__ __forceinline__ int regid(int p) { return p >= 252 ? 2 : (p >= 248 ? 1 : 0); }

// CDNA5 async global->LDS copy (ASYNCcnt-tracked), 16B per lane.
static __device__ __forceinline__ void async_copy_b128(unsigned lds_off, const void* gptr) {
  asm volatile("global_load_async_to_lds_b128 %0, %1, off"
               :: "v"(lds_off), "v"((unsigned long long)(uintptr_t)gptr)
               : "memory");
}
static __device__ __forceinline__ void async_wait0() {
  asm volatile("s_wait_asynccnt 0x0" ::: "memory");
}

// ---------------- prologue: LDS-tiled NCHW -> NHWC transpose (both sides coalesced) ----------------
__global__ __launch_bounds__(256) void nchw_to_nhwc(const float* __restrict__ in,
                                                    float* __restrict__ out) {
  __shared__ float tile[32][33];
  int bid = blockIdx.x;               // 2*256*4*8 = 16384 blocks
  int wg  = bid & 7;
  int cg  = (bid >> 3) & 3;
  int h   = (bid >> 5) & 255;
  int b   = bid >> 13;
  int lx  = threadIdx.x & 31;
  int ly  = threadIdx.x >> 5;         // 0..7
#pragma unroll
  for (int it = 0; it < 4; ++it) {    // read: lanes sweep w (contiguous)
    int c = cg * 32 + ly + it * 8;
    tile[ly + it * 8][lx] = in[(((size_t)b * C_DIM + c) * IMG + h) * IMG + wg * 32 + lx];
  }
  __syncthreads();
#pragma unroll
  for (int it = 0; it < 4; ++it) {    // write: lanes sweep c (contiguous)
    int w = wg * 32 + ly + it * 8;
    out[(((size_t)b * IMG + h) * IMG + w) * C_DIM + cg * 32 + lx] = tile[lx][ly + it * 8];
  }
}

__global__ void cvt_f32_f16(const float* __restrict__ s, f16* __restrict__ d, int n) {
  int i = blockIdx.x * 256 + threadIdx.x;
  if (i < n) d[i] = (f16)s[i];
}

// ---------------- fused window attention (one 8x8 window per block) ----------------
__global__ __launch_bounds__(256) void swin_attn(
    const float* __restrict__ in, float* __restrict__ out,
    const float* __restrict__ ng, const float* __restrict__ nb,
    const f16* __restrict__ qkvw, const float* __restrict__ qkvb,
    const f16* __restrict__ projw, const float* __restrict__ projb,
    const float* __restrict__ rpb, int shift)
{
  extern __shared__ char smem[];
  float* xw   = (float*)smem;                                  // [64][128] f32 (residual)
  int*   gpos = (int*)  (smem + NTOK * C_DIM * 4);             // [64]
  f16*   xn   = (f16*)  (smem + NTOK * C_DIM * 4 + 256);       // [64][128] LN'ed
  f16*   qkvh = (f16*)  ((char*)xn   + NTOK * C_DIM * 2);      // [64][384] (q pre-scaled)
  float* sbuf = (float*)((char*)qkvh + NTOK * 384 * 2);        // [64][64] scores
  f16*   pbuf = (f16*)  ((char*)sbuf + NTOK * 64 * 4);         // [64][64] probs
  f16*   obuf = (f16*)  ((char*)pbuf + NTOK * 64 * 2);         // [64][128] attn out

  const int tid  = threadIdx.x;
  const int lane = tid & 31;
  const int wave = tid >> 5;

  const int wid  = blockIdx.x;
  const int b    = wid >> 10;
  const int wh   = (wid >> 5) & 31;
  const int wc   = wid & 31;

  // ---- async global->LDS window staging (roll folded into gather index) ----
  const unsigned xw_base = (unsigned)(uintptr_t)xw;
  for (int t = wave; t < NTOK; t += 8) {
    int r = t >> 3, c = t & 7;
    int h = wh * 8 + r, w = wc * 8 + c;
    if (shift) { h = (h + 4) & 255; w = (w + 4) & 255; }
    int pos = (b * IMG + h) * IMG + w;
    if (lane == 0) gpos[t] = pos;
    async_copy_b128(xw_base + (unsigned)(t * 512 + lane * 16),
                    in + (size_t)pos * C_DIM + lane * 4);   // 32 lanes x 16B = one token row
  }
  async_wait0();
  __syncthreads();

  // ---- LayerNorm -> f16 ----
  if (tid < NTOK) {
    float mu = 0.f;
    for (int i = 0; i < C_DIM; ++i) mu += xw[tid * C_DIM + i];
    mu *= (1.f / C_DIM);
    float var = 0.f;
    for (int i = 0; i < C_DIM; ++i) { float d = xw[tid * C_DIM + i] - mu; var += d * d; }
    float rs = rsqrtf(var * (1.f / C_DIM) + 1e-5f);
    for (int i = 0; i < C_DIM; ++i)
      xn[tid * C_DIM + i] = (f16)((xw[tid * C_DIM + i] - mu) * rs * ng[i] + nb[i]);
  }
  __syncthreads();

  // ---- QKV GEMM: [64x128] x [128x384] (W^T), q scaled by hd^-0.5 ----
  for (int tile = wave; tile < 96; tile += 8) {
    int mt = tile / 24, nt = tile % 24;
    v8f acc = {0.f,0.f,0.f,0.f,0.f,0.f,0.f,0.f};
#pragma unroll
    for (int kk = 0; kk < 4; ++kk) {
      v16h a  = load_a (xn,   C_DIM, mt * 16, kk * 32, lane);
      v16h bb = load_bT(qkvw, C_DIM, nt * 16, kk * 32, lane);
      acc = wmma_f16(a, bb, acc);
    }
    int n = lane & 15, mb = (lane >> 4) * 8;
    int col = nt * 16 + n;
    float bias = qkvb[col];
    float sc = (col < C_DIM) ? 0.17677669529663687f : 1.0f;   // 1/sqrt(32) on q
#pragma unroll
    for (int r = 0; r < 8; ++r)
      qkvh[(mt * 16 + mb + r) * 384 + col] = (f16)((acc[r] + bias) * sc);
  }
  __syncthreads();

  // ---- per-head attention ----
  for (int hd = 0; hd < NHEAD; ++hd) {
    for (int tile = wave; tile < 16; tile += 8) {      // S = q @ k^T (K=32)
      int mt = tile >> 2, nt = tile & 3;
      v16h a  = load_a (qkvh, 384, mt * 16, hd * HD, lane);
      v16h bb = load_bT(qkvh, 384, nt * 16, C_DIM + hd * HD, lane);
      v8f acc = {0.f,0.f,0.f,0.f,0.f,0.f,0.f,0.f};
      acc = wmma_f16(a, bb, acc);
      int n = lane & 15, mb = (lane >> 4) * 8;
#pragma unroll
      for (int r = 0; r < 8; ++r) sbuf[(mt * 16 + mb + r) * 64 + nt * 16 + n] = acc[r];
    }
    __syncthreads();

    if (tid < NTOK) {                                  // bias + mask + softmax
      int i = tid, ri = i >> 3, ci = i & 7;
      int gi = shift ? regid(wh * 8 + ri) * 3 + regid(wc * 8 + ci) : 0;
      float mx = -1e30f;
      for (int j = 0; j < 64; ++j) {
        int rj = j >> 3, cj = j & 7;
        int idx = (ri - rj + 7) * 15 + (ci - cj + 7);
        float v = sbuf[i * 64 + j] + rpb[idx * NHEAD + hd];
        if (shift) {
          int gj = regid(wh * 8 + rj) * 3 + regid(wc * 8 + cj);
          if (gj != gi) v -= 100.f;
        }
        sbuf[i * 64 + j] = v;
        mx = fmaxf(mx, v);
      }
      float sum = 0.f;
      for (int j = 0; j < 64; ++j) {
        float e = expf(sbuf[i * 64 + j] - mx);
        sbuf[i * 64 + j] = e;
        sum += e;
      }
      float inv = 1.f / sum;
      for (int j = 0; j < 64; ++j) pbuf[i * 64 + j] = (f16)(sbuf[i * 64 + j] * inv);
    }
    __syncthreads();

    {                                                  // O_h = P @ V_h, one tile/wave
      int mt = wave >> 1, nt = wave & 1;
      v8f acc = {0.f,0.f,0.f,0.f,0.f,0.f,0.f,0.f};
#pragma unroll
      for (int kk = 0; kk < 2; ++kk) {
        v16h a  = load_a(pbuf, 64, mt * 16, kk * 32, lane);
        v16h bb = load_b(qkvh, 384, kk * 32, 2 * C_DIM + hd * HD + nt * 16, lane);
        acc = wmma_f16(a, bb, acc);
      }
      int n = lane & 15, mb = (lane >> 4) * 8;
#pragma unroll
      for (int r = 0; r < 8; ++r)
        obuf[(mt * 16 + mb + r) * C_DIM + hd * HD + nt * 16 + n] = (f16)acc[r];
    }
    __syncthreads();
  }

  // ---- proj + bias + residual -> global ----
  for (int tile = wave; tile < 32; tile += 8) {
    int mt = tile >> 3, nt = tile & 7;
    v8f acc = {0.f,0.f,0.f,0.f,0.f,0.f,0.f,0.f};
#pragma unroll
    for (int kk = 0; kk < 4; ++kk) {
      v16h a  = load_a (obuf,  C_DIM, mt * 16, kk * 32, lane);
      v16h bb = load_bT(projw, C_DIM, nt * 16, kk * 32, lane);
      acc = wmma_f16(a, bb, acc);
    }
    int n = lane & 15, mb = (lane >> 4) * 8;
    int col = nt * 16 + n;
    float pb = projb[col];
#pragma unroll
    for (int r = 0; r < 8; ++r) {
      int m = mt * 16 + mb + r;
      out[(size_t)gpos[m] * C_DIM + col] = acc[r] + pb + xw[m * C_DIM + col];
    }
  }
}

// ---------------- fused FF: LN -> 1x1 (128->512) -> GELU -> dw3x3 -> GELU -> 1x1 (512->128) ----------------
// smem: h1 [112][512] f16 | region R (33,024 B): {xn [112][128] f16, g2 [64][32] f16} then
//       reused as otile [64][129] f32 for the coalesced NCHW store.
__global__ __launch_bounds__(256) void swin_ff(
    const float* __restrict__ in,  // [B,H,W,C] f32 (also residual)
    float* __restrict__ out,       // [B,C,H,W] f32 final output
    const float* __restrict__ ng, const float* __restrict__ nb,
    const f16* __restrict__ w1,    // [512][128] f16
    const float* __restrict__ dw,  // [3][3][512] f32
    const f16* __restrict__ w2)    // [128][512] f16
{
  extern __shared__ char smem[];
  f16*   h1    = (f16*)smem;                                   // 114,688 B
  f16*   xn    = (f16*)(smem + 112 * M_DIM * 2);               //  28,672 B
  f16*   g2    = (f16*)(smem + 112 * M_DIM * 2 + 112 * C_DIM * 2); // 4,096 B
  float* otile = (float*)(smem + 112 * M_DIM * 2);             // reuses xn+g2, ld=129

  const int tid  = threadIdx.x;
  const int lane = tid & 31;
  const int wave = tid >> 5;

  const int wid = blockIdx.x;
  const int b   = wid >> 10;
  const int th  = (wid >> 5) & 31;
  const int tw  = wid & 31;

  // ---- LN of 10x10 halo (zero outside image => zero-padded conv input) ----
  if (tid < 112) {
    bool valid = false;
    if (tid < 100) {
      int hr = tid / 10, hc = tid % 10;
      int h = th * 8 + hr - 1, w = tw * 8 + hc - 1;
      if (h >= 0 && h < IMG && w >= 0 && w < IMG) {
        valid = true;
        const float* src = in + ((size_t)(b * IMG + h) * IMG + w) * C_DIM;
        float mu = 0.f;
        for (int i = 0; i < C_DIM; ++i) mu += src[i];
        mu *= (1.f / C_DIM);
        float var = 0.f;
        for (int i = 0; i < C_DIM; ++i) { float d = src[i] - mu; var += d * d; }
        float rs = rsqrtf(var * (1.f / C_DIM) + 1e-5f);
        for (int i = 0; i < C_DIM; ++i)
          xn[tid * C_DIM + i] = (f16)((src[i] - mu) * rs * ng[i] + nb[i]);
      }
    }
    if (!valid)
      for (int i = 0; i < C_DIM; ++i) xn[tid * C_DIM + i] = (f16)0.f;
  }
  __syncthreads();

  // ---- h1 = gelu(xn @ W1^T): [112x128] x [128x512] ----
  for (int tile = wave; tile < 7 * 32; tile += 8) {
    int mt = tile >> 5, nt = tile & 31;
    v8f acc = {0.f,0.f,0.f,0.f,0.f,0.f,0.f,0.f};
#pragma unroll
    for (int kk = 0; kk < 4; ++kk) {
      v16h a  = load_a (xn, C_DIM, mt * 16, kk * 32, lane);
      v16h bb = load_bT(w1, C_DIM, nt * 16, kk * 32, lane);
      acc = wmma_f16(a, bb, acc);
    }
    int n = lane & 15, mb = (lane >> 4) * 8;
#pragma unroll
    for (int r = 0; r < 8; ++r)
      h1[(mt * 16 + mb + r) * M_DIM + nt * 16 + n] = (f16)gelu_f(acc[r]);
  }

  // ---- K-chunked: dw3x3 + gelu into g2, immediately consumed by 512->128 GEMM ----
  const int mt  = wave >> 1;
  const int nc0 = (wave & 1) * 64;
  v8f acc4[4];
#pragma unroll
  for (int j = 0; j < 4; ++j) acc4[j] = (v8f){0.f,0.f,0.f,0.f,0.f,0.f,0.f,0.f};

  for (int kc = 0; kc < 16; ++kc) {
    __syncthreads();                       // h1 ready / previous g2 consumed
    for (int item = tid; item < 64 * 32; item += 256) {
      int tok = item >> 5, cc = item & 31;
      int ch = kc * 32 + cc;
      int r = tok >> 3, c = tok & 7;
      float s = 0.f;
#pragma unroll
      for (int dy = 0; dy < 3; ++dy)
#pragma unroll
        for (int dx = 0; dx < 3; ++dx)
          s += (float)h1[((r + dy) * 10 + (c + dx)) * M_DIM + ch] * dw[(dy * 3 + dx) * M_DIM + ch];
      g2[tok * 32 + cc] = (f16)gelu_f(s);
    }
    __syncthreads();
    v16h a = load_a(g2, 32, mt * 16, 0, lane);
#pragma unroll
    for (int j = 0; j < 4; ++j) {
      v16h bb = load_bT(w2, M_DIM, nc0 + j * 16, kc * 32, lane);
      acc4[j] = wmma_f16(a, bb, acc4[j]);
    }
  }
  __syncthreads();                         // last GEMM done reading g2 region

  // ---- residual add (coalesced NHWC read) into LDS tile ----
  {
    int n = lane & 15, mb = (lane >> 4) * 8;
#pragma unroll
    for (int j = 0; j < 4; ++j) {
      int col = nc0 + j * 16 + n;
#pragma unroll
      for (int r = 0; r < 8; ++r) {
        int m = mt * 16 + mb + r;
        int hh = th * 8 + (m >> 3), ww = tw * 8 + (m & 7);
        float res = in[((size_t)(b * IMG + hh) * IMG + ww) * C_DIM + col];
        otile[m * 129 + col] = acc4[j][r] + res;
      }
    }
  }
  __syncthreads();

  // ---- NCHW-linear store: 8-element contiguous runs per (c,h) row ----
  for (int idx = tid; idx < 64 * C_DIM; idx += 256) {
    int col = idx >> 6;                    // channel
    int m   = idx & 63;                    // token (r*8+c)
    int hh = th * 8 + (m >> 3), ww = tw * 8 + (m & 7);
    out[(((size_t)b * C_DIM + col) * IMG + hh) * IMG + ww] = otile[m * 129 + col];
  }
}

extern "C" void kernel_launch(void* const* d_in, const int* in_sizes, int n_in,
                              void* d_out, int out_size, void* d_ws, size_t ws_size,
                              hipStream_t stream) {
  const float* x        = (const float*)d_in[0];
  const float* w_ng     = (const float*)d_in[1];
  const float* w_nb     = (const float*)d_in[2];
  const float* w_qkv_w  = (const float*)d_in[3];
  const float* w_qkv_b  = (const float*)d_in[4];
  const float* w_proj_w = (const float*)d_in[5];
  const float* w_proj_b = (const float*)d_in[6];
  const float* w_rpb    = (const float*)d_in[7];
  const float* s_ng     = (const float*)d_in[8];
  const float* s_nb     = (const float*)d_in[9];
  const float* s_qkv_w  = (const float*)d_in[10];
  const float* s_qkv_b  = (const float*)d_in[11];
  const float* s_proj_w = (const float*)d_in[12];
  const float* s_proj_b = (const float*)d_in[13];
  const float* s_rpb    = (const float*)d_in[14];
  const float* ff_ng    = (const float*)d_in[15];
  const float* ff_nb    = (const float*)d_in[16];
  const float* ff_w1    = (const float*)d_in[17];
  const float* ff_dw    = (const float*)d_in[18];
  const float* ff_w2    = (const float*)d_in[19];

  char* ws = (char*)d_ws;
  const size_t TENSOR_BYTES = (size_t)2 * IMG * IMG * C_DIM * 4;   // 67,108,864
  float* bufA = (float*)ws;                          // xT, later y2
  float* bufB = (float*)(ws + TENSOR_BYTES);         // y1
  f16* qkv1h  = (f16*)(ws + 2 * TENSOR_BYTES);
  f16* proj1h = qkv1h  + 384 * 128;
  f16* qkv2h  = proj1h + 128 * 128;
  f16* proj2h = qkv2h  + 384 * 128;
  f16* w1h    = proj2h + 128 * 128;
  f16* w2h    = w1h    + 512 * 128;

  nchw_to_nhwc<<<16384, 256, 0, stream>>>(x, bufA);
  cvt_f32_f16<<<192, 256, 0, stream>>>(w_qkv_w,  qkv1h,  384 * 128);
  cvt_f32_f16<<< 64, 256, 0, stream>>>(w_proj_w, proj1h, 128 * 128);
  cvt_f32_f16<<<192, 256, 0, stream>>>(s_qkv_w,  qkv2h,  384 * 128);
  cvt_f32_f16<<< 64, 256, 0, stream>>>(s_proj_w, proj2h, 128 * 128);
  cvt_f32_f16<<<256, 256, 0, stream>>>(ff_w1,    w1h,    512 * 128);
  cvt_f32_f16<<<256, 256, 0, stream>>>(ff_w2,    w2h,    128 * 512);

  const size_t attn_smem = NTOK*C_DIM*4 + 256 + NTOK*C_DIM*2 + NTOK*384*2
                         + NTOK*64*4 + NTOK*64*2 + NTOK*C_DIM*2;   // 139,520 B
  swin_attn<<<2048, 256, attn_smem, stream>>>(bufA, bufB, w_ng, w_nb, qkv1h, w_qkv_b,
                                              proj1h, w_proj_b, w_rpb, 0);
  swin_attn<<<2048, 256, attn_smem, stream>>>(bufB, bufA, s_ng, s_nb, qkv2h, s_qkv_b,
                                              proj2h, s_proj_b, s_rpb, 1);

  const size_t ff_smem = 112*M_DIM*2 + 64*129*4;                    // 114,688 + 33,024 B
  swin_ff<<<2048, 256, ff_smem, stream>>>(bufA, (float*)d_out, ff_ng, ff_nb, w1h, ff_dw, w2h);

  (void)in_sizes; (void)n_in; (void)out_size; (void)ws_size;
}
